// POI_SSL_16466904613034
// MI455X (gfx1250) — matrix operand order repaired and checked by reading, not compiled, not run on hardware
//
#include <hip/hip_runtime.h>
#include <hip/hip_bf16.h>
#include <math.h>

#define EMB 128
#define ATT 32
#define POI_NUM 500000
#define WPB 4        // waves per block (blockDim = 128, wave32)
#define GROUP 16     // parents per wave-group (WMMA M dim)

typedef __attribute__((ext_vector_type(16))) __bf16 v16bf;
typedef __attribute__((ext_vector_type(8)))  float  v8f;
typedef __attribute__((ext_vector_type(4)))  float  f32x4;
typedef __attribute__((ext_vector_type(4)))  unsigned int u32x4;

union FragBf { v16bf v; u32x4 q[2]; };

// A-matrix fragment (16xK bf16 tile, row-major in LDS as [16][128] bf16).
// ISA 7.12.2: lanes 0-15 row M=lane hold K = kt*32 + {0..7,16..23};
//             lanes 16-31 row M=lane-16 hold K = kt*32 + {8..15,24..31}.
static __device__ __forceinline__ FragBf load_a_frag(const __bf16* m, int lane, int kt) {
    int row = lane & 15;
    int kb  = kt * 32 + ((lane >> 4) << 3);
    FragBf f;
    f.q[0] = *(const u32x4*)(m + row * EMB + kb);
    f.q[1] = *(const u32x4*)(m + row * EMB + kb + 16);
    return f;
}

// B-matrix fragment from W^T stored as [32 cols][128 K] bf16 in LDS.
// lanes 0-15: col n = nt*16+lane, K = kt*32 + {0..7,16..23};
// lanes 16-31: col n = nt*16+lane-16, K = kt*32 + {8..15,24..31}.
static __device__ __forceinline__ FragBf load_b_frag(const __bf16* wT, int lane, int kt, int nt) {
    int n  = nt * 16 + (lane & 15);
    int ko = kt * 32 + ((lane >> 4) << 3);
    FragBf f;
    f.q[0] = *(const u32x4*)(wT + n * EMB + ko);
    f.q[1] = *(const u32x4*)(wT + n * EMB + ko + 16);
    return f;
}

static __device__ __forceinline__ void store_bf4(__bf16* d, f32x4 g) {
    d[0] = (__bf16)g[0]; d[1] = (__bf16)g[1]; d[2] = (__bf16)g[2]; d[3] = (__bf16)g[3];
}

// Hardware transcendentals (CDNA5 has V_TANH_F32 / V_EXP_F32 TRANS ops that
// co-execute with the matrix pipe; avoid libm software expansions).
static __device__ __forceinline__ float fast_tanh(float x) {
#if __has_builtin(__builtin_amdgcn_tanhf)
    return __builtin_amdgcn_tanhf(x);
#else
    float e = __expf(2.0f * x);
    return 1.0f - 2.0f * __builtin_amdgcn_rcpf(e + 1.0f);
#endif
}

// One hierarchy level: for each parent p in [0,P):
//   first vector  = firstRows[p]                (same for all fan slots)
//   second rows   = table[idx[p, j]]            (redirected to `prev` if idx in write range)
//   h   = tanh([first,second] @ W_att + b_att)
//   pre = h . v_att + mask ; att = softmax(pre over fan)
//   emb[off + p] = sum_j att[j] * second[j]
__global__ void __launch_bounds__(WPB * 32)
poi_level_kernel(float* __restrict__ emb,
                 const float* __restrict__ prev,
                 const float* __restrict__ firstRows,
                 const int*   __restrict__ idxp,
                 const float* __restrict__ maskp,
                 const float* __restrict__ W_att,
                 const float* __restrict__ b_att,
                 const float* __restrict__ v_att,
                 int off, int P, int fan) {
    __shared__ __align__(16) __bf16 sW1T[ATT * EMB];           // first-half  W_att^T (bf16)
    __shared__ __align__(16) __bf16 sW2T[ATT * EMB];           // second-half W_att^T (bf16)
    __shared__ float sB[ATT];
    __shared__ float sV[ATT];
    __shared__ __align__(16) __bf16 sF [WPB][GROUP * EMB];     // 16 parent first-vectors (bf16)
    __shared__ __align__(16) __bf16 sA [WPB][GROUP * EMB];     // gathered child rows (bf16, WMMA A)
    __shared__ __align__(16) float  sWc[WPB][GROUP * EMB];     // gathered child rows (fp32, output)
    __shared__ float sApre[WPB][GROUP * ATT];                  // first-vector @ W1 per group

    const int tid = threadIdx.x;
    // Stage W_att transposed halves as bf16 (B-operand friendly layout).
    for (int t = tid; t < ATT * EMB; t += blockDim.x) {
        int n = t >> 7, k = t & 127;
        sW1T[t] = (__bf16)W_att[(size_t)k * ATT + n];
        sW2T[t] = (__bf16)W_att[(size_t)(k + EMB) * ATT + n];
    }
    if (tid < ATT) { sB[tid] = b_att[tid]; sV[tid] = v_att[tid]; }
    __syncthreads();

    const int wv   = tid >> 5;
    const int lane = tid & 31;
    const int half = lane >> 4;
    const int l16  = lane & 15;
    const int groups = (P + GROUP - 1) / GROUP;

    // W2 is constant for the whole kernel: hoist its 8 B-fragments into
    // registers (64 VGPRs) so the hot per-parent loop only pulls A-fragments
    // from LDS (8 x ds_load_b128 per parent instead of 24).
    FragBf bw2[4][2];
#pragma unroll
    for (int kt = 0; kt < 4; ++kt) {
        bw2[kt][0] = load_b_frag(sW2T, lane, kt, 0);
        bw2[kt][1] = load_b_frag(sW2T, lane, kt, 1);
    }

    for (int g = blockIdx.x * WPB + wv; g < groups; g += gridDim.x * WPB) {
        const int p0 = g * GROUP;

        // ---- Stage the 16 parent "first" vectors, compute Apre = F @ W1 via WMMA ----
        for (int jj = 0; jj < GROUP; ++jj) {
            int pg = p0 + jj; if (pg > P - 1) pg = P - 1;
            f32x4 gv = ((const f32x4*)(firstRows + (size_t)pg * EMB))[lane];
            store_bf4(sF[wv] + jj * EMB + 4 * lane, gv);
        }
        __builtin_amdgcn_wave_barrier();

        v8f accA0 = {}, accA1 = {};
#pragma unroll
        for (int kt = 0; kt < 4; ++kt) {
            FragBf a  = load_a_frag(sF[wv], lane, kt);
            FragBf b0 = load_b_frag(sW1T, lane, kt, 0);
            FragBf b1 = load_b_frag(sW1T, lane, kt, 1);
            accA0 = __builtin_amdgcn_wmma_f32_16x16x32_bf16(false, a.v, false, b0.v, (short)0, accA0, false, false);
            accA1 = __builtin_amdgcn_wmma_f32_16x16x32_bf16(false, a.v, false, b1.v, (short)0, accA1, false, false);
        }
#pragma unroll
        for (int r = 0; r < 8; ++r) {
            sApre[wv][(r + half * 8) * ATT + l16]      = accA0[r];
            sApre[wv][(r + half * 8) * ATT + 16 + l16] = accA1[r];
        }
        __builtin_amdgcn_wave_barrier();

        // ---- Per-parent attention ----
        for (int pl = 0; pl < GROUP; ++pl) {
            const int p = p0 + pl;
            if (p >= P) break;

            // Gather fan child rows (pad to 16 with zeros); keep fp32 + bf16 copies.
            for (int j = 0; j < GROUP; ++j) {
                f32x4 gv = {0.f, 0.f, 0.f, 0.f};
                if (j < fan) {
                    int idx = idxp[(size_t)p * fan + j];
                    const float* src = (idx >= off && idx < off + P)
                                         ? prev + (size_t)(idx - off) * EMB
                                         : emb  + (size_t)idx * EMB;
                    gv = ((const f32x4*)src)[lane];
                }
                ((f32x4*)(sWc[wv] + j * EMB))[lane] = gv;
                store_bf4(sA[wv] + j * EMB + 4 * lane, gv);
            }
            __builtin_amdgcn_wave_barrier();

            // H2 = Wc @ W2 : [16 x 32] via 8 WMMAs (2 N-tiles x 4 K-tiles).
            v8f acc0 = {}, acc1 = {};
#pragma unroll
            for (int kt = 0; kt < 4; ++kt) {
                FragBf a = load_a_frag(sA[wv], lane, kt);
                acc0 = __builtin_amdgcn_wmma_f32_16x16x32_bf16(false, a.v, false, bw2[kt][0].v, (short)0, acc0, false, false);
                acc1 = __builtin_amdgcn_wmma_f32_16x16x32_bf16(false, a.v, false, bw2[kt][1].v, (short)0, acc1, false, false);
            }

            // h = tanh(H2 + Apre + b);  t = h * v;  pre[row] = sum over 32 att cols.
            const float apre0 = sApre[wv][pl * ATT + l16];
            const float apre1 = sApre[wv][pl * ATT + 16 + l16];
            const float bb0 = sB[l16], bb1 = sB[16 + l16];
            const float vv0 = sV[l16], vv1 = sV[16 + l16];
            float s[8];
#pragma unroll
            for (int r = 0; r < 8; ++r) {
                float h0 = fast_tanh(acc0[r] + apre0 + bb0);
                float h1 = fast_tanh(acc1[r] + apre1 + bb1);
                s[r] = h0 * vv0 + h1 * vv1;
            }
            // Reduce across the 16 lanes of each half (D layout: col = lane%16).
#pragma unroll
            for (int r = 0; r < 8; ++r) {
                s[r] += __shfl_xor(s[r], 1, 32);
                s[r] += __shfl_xor(s[r], 2, 32);
                s[r] += __shfl_xor(s[r], 4, 32);
                s[r] += __shfl_xor(s[r], 8, 32);
                int row = r + half * 8;
                s[r] += (row < fan) ? maskp[(size_t)p * fan + row] : -1e13f;
            }
            // Exchange halves so every lane holds all 16 pre values.
            float pre_lo[8], pre_hi[8];
#pragma unroll
            for (int r = 0; r < 8; ++r) {
                float o = __shfl_xor(s[r], 16, 32);
                pre_lo[r] = half ? o : s[r];
                pre_hi[r] = half ? s[r] : o;
            }
            float mx = -3.4e38f;
#pragma unroll
            for (int r = 0; r < 8; ++r) mx = fmaxf(mx, fmaxf(pre_lo[r], pre_hi[r]));
            float wl[8], wh[8], Z = 0.f;
#pragma unroll
            for (int r = 0; r < 8; ++r) {
                wl[r] = __expf(pre_lo[r] - mx); Z += wl[r];
                wh[r] = __expf(pre_hi[r] - mx); Z += wh[r];
            }
            const float inv = __builtin_amdgcn_rcpf(Z);

            // out = sum_c att[c] * Wc[c]  (fp32), coalesced float4 store.
            f32x4 outv = {0.f, 0.f, 0.f, 0.f};
#pragma unroll
            for (int c = 0; c < GROUP; ++c) {
                float w = ((c < 8) ? wl[c] : wh[c - 8]) * inv;
                f32x4 e = ((const f32x4*)(sWc[wv] + c * EMB))[lane];
                outv[0] += w * e[0]; outv[1] += w * e[1];
                outv[2] += w * e[2]; outv[3] += w * e[3];
            }
            ((f32x4*)(emb + (size_t)(off + p) * EMB))[lane] = outv;
        }
    }
}

extern "C" void kernel_launch(void* const* d_in, const int* in_sizes, int n_in,
                              void* d_out, int out_size, void* d_ws, size_t ws_size,
                              hipStream_t stream) {
    (void)in_sizes; (void)n_in; (void)out_size; (void)ws_size;
    const float* W_poi = (const float*)d_in[0];
    const float* W_att = (const float*)d_in[1];
    const float* b_att = (const float*)d_in[2];
    const float* v_att = (const float*)d_in[3];
    float* emb  = (float*)d_out;
    float* prev = (float*)d_ws;   // snapshot of the slice being overwritten (max 25.6 MB)

    // W_emb = W_poi.clone()
    hipMemcpyAsync(emb, W_poi, (size_t)POI_NUM * EMB * sizeof(float),
                   hipMemcpyDeviceToDevice, stream);

    const int offs[3] = {0, 50000, 55000};
    const int Ps[3]   = {50000, 5000, 500};

    for (int lvl = 0; lvl < 6; ++lvl) {
        const int i   = lvl % 3;
        const bool c2p = (lvl >= 3);
        const int off = offs[i], P = Ps[i];
        const int fan = c2p ? 8 : 16;
        const int* idx    = c2p ? (const int*)d_in[14 + 3 * i] : (const int*)d_in[5 + 3 * i];
        const float* mask = c2p ? (const float*)d_in[15 + 3 * i] : (const float*)d_in[6 + 3 * i];

        // Snapshot the write range so all intra-level gathers see pre-level values.
        hipMemcpyAsync(prev, emb + (size_t)off * EMB, (size_t)P * EMB * sizeof(float),
                       hipMemcpyDeviceToDevice, stream);

        // p2c: first vector = W_poi row (reference gathers parents from original W_poi).
        // c2p: first vector = the row's own pre-level value (= snapshot row p).
        const float* firstRows = c2p ? prev : (W_poi + (size_t)off * EMB);

        int groups = (P + GROUP - 1) / GROUP;
        int blocks = (groups + WPB - 1) / WPB;
        if (blocks > 8192) blocks = 8192;
        poi_level_kernel<<<blocks, WPB * 32, 0, stream>>>(
            emb, prev, firstRows, idx, mask, W_att, b_att, v_att, off, P, fan);
    }
}
